// Backflow_79809082294809
// MI455X (gfx1250) — matrix operand order repaired and checked by reading, not compiled, not run on hardware
//
#include <hip/hip_runtime.h>

typedef __attribute__((ext_vector_type(16))) _Float16 v16h;
typedef __attribute__((ext_vector_type(8)))  _Float16 v8h;
typedef __attribute__((ext_vector_type(8)))  float    v8f;

#define LN2F 0.69314718055994530942f

__device__ __forceinline__ float ssp(float x) {
    // log(0.5*exp(x)+0.5) = softplus(x) - ln2, numerically stable
    return fmaxf(x, 0.0f) + log1pf(expf(-fabsf(x))) - LN2F;
}

__device__ __forceinline__ v8f wmma16(v16h a, v16h b, v8f c) {
    return __builtin_amdgcn_wmma_f32_16x16x32_f16(false, a, false, b, (short)0, c,
                                                  false, false);
}

__device__ __forceinline__ void wave_sync_lds() {
    __builtin_amdgcn_fence(__ATOMIC_ACQ_REL, "wavefront");
    __builtin_amdgcn_wave_barrier();
}

// ---------------------------------------------------------------------------
// Kernel A: electron-nucleus path.
// rows = B*N = 16384 -> 1024 M-tiles of 16 rows. Block = 64 threads (2 waves),
// one tile per wave, grid = 512. GEMM [16,256]x[256,81->96 pad] via WMMA f16.
// Writes bf_nuc[row][3] and cutoff[row] to workspace.
// ---------------------------------------------------------------------------
__global__ __launch_bounds__(64) void nuc_kernel(
    const float* __restrict__ rs, const float* __restrict__ xs,
    const float* __restrict__ coords,
    const float* __restrict__ nw1, const float* __restrict__ nb1,
    const float* __restrict__ nw2, const float* __restrict__ nb2,
    const float* __restrict__ nw3, const float* __restrict__ nb3,
    float* __restrict__ bfnuc_ws, float* __restrict__ cut_ws)
{
    __shared__ __align__(32) _Float16 w1f[3 * 8 * 32 * 16]; // half of B frags, 24KB
    __shared__ float scr1[2][16 * 84];
    __shared__ float scr2[2][16 * 28];
    __shared__ float scr3[2][16 * 8];
    __shared__ float s_b1[96];
    __shared__ float s_w2[81 * 25];
    __shared__ float s_b2[25];
    __shared__ float s_w3[25 * 8];
    __shared__ float s_b3[8];
    __shared__ float s_coords[24];

    const int tid  = threadIdx.x;
    const int lane = tid & 31;
    const int wave = tid >> 5;
    const bool hi  = lane >= 16;
    const int mrow = lane & 15;
    const int a_off0 = hi ? 8 : 0;
    const int tile = blockIdx.x * 2 + wave;
    const int R = tile * 16 + mrow;
    const float* xr = xs + (size_t)R * 256;

    for (int i = tid; i < 96; i += 64)      s_b1[i] = (i < 81) ? nb1[i] : 0.0f;
    for (int i = tid; i < 81 * 25; i += 64) s_w2[i] = nw2[i];
    if (tid < 25) s_b2[tid] = nb2[tid];
    for (int i = tid; i < 200; i += 64)     s_w3[i] = nw3[i];
    if (tid < 8)  s_b3[tid] = nb3[tid];
    if (tid < 24) s_coords[tid] = coords[tid];

    v8f acc[6] = {};

    for (int half = 0; half < 2; ++half) {
        __syncthreads();
        // stage nw1 fragments for N-tiles [half*3, half*3+3)
        for (int idx = tid; idx < 3 * 8 * 32; idx += 64) {
            int l = idx & 31;
            int s = (idx >> 5) & 7;
            int t = idx >> 8;
            int n  = (half * 3 + t) * 16 + (l & 15);
        int kb = s * 32 + ((l >= 16) ? 8 : 0);
            _Float16* dst = &w1f[idx * 16];
            #pragma unroll
            for (int e = 0; e < 16; ++e)
                dst[e] = (n < 81) ? (_Float16)nw1[(kb + e + ((e >= 8) ? 8 : 0)) * 81 + n] : (_Float16)0.0f;
        }
        __syncthreads();

        const v16h* bfr = (const v16h*)w1f;
        for (int s = 0; s < 8; ++s) {
            int k0 = s * 32 + a_off0;
            float4 f0 = *(const float4*)(xr + k0);
            float4 f1 = *(const float4*)(xr + k0 + 4);
            float4 f2 = *(const float4*)(xr + k0 + 16);
            float4 f3 = *(const float4*)(xr + k0 + 20);
            v16h a;
            a[0]=(_Float16)f0.x;  a[1]=(_Float16)f0.y;  a[2]=(_Float16)f0.z;  a[3]=(_Float16)f0.w;
            a[4]=(_Float16)f1.x;  a[5]=(_Float16)f1.y;  a[6]=(_Float16)f1.z;  a[7]=(_Float16)f1.w;
            a[8]=(_Float16)f2.x;  a[9]=(_Float16)f2.y;  a[10]=(_Float16)f2.z; a[11]=(_Float16)f2.w;
            a[12]=(_Float16)f3.x; a[13]=(_Float16)f3.y; a[14]=(_Float16)f3.z; a[15]=(_Float16)f3.w;
            #pragma unroll
            for (int t = 0; t < 3; ++t) {
                v16h bm = bfr[(t * 8 + s) * 32 + lane];
                acc[half * 3 + t] = wmma16(a, bm, acc[half * 3 + t]);
            }
        }
    }

    // h1 = ssp(acc + b1) -> per-wave scratch [16 rows][81] (stride 84)
    {
        float* sc = scr1[wave];
        int mb = hi ? 8 : 0;
        int nl = lane & 15;
        #pragma unroll
        for (int t = 0; t < 6; ++t) {
            int n = t * 16 + nl;
            if (n < 81) {
                float bias = s_b1[n];
                #pragma unroll
                for (int r = 0; r < 8; ++r)
                    sc[(mb + r) * 84 + n] = ssp(acc[t][r] + bias);
            }
        }
    }
    wave_sync_lds();

    // layer 2: 81 -> 25, two lanes per row
    {
        const float* hrow = &scr1[wave][mrow * 84];
        for (int c = hi ? 1 : 0; c < 25; c += 2) {
            float a2 = s_b2[c];
            for (int n = 0; n < 81; ++n) a2 += hrow[n] * s_w2[n * 25 + c];
            scr2[wave][mrow * 28 + c] = ssp(a2);
        }
    }
    wave_sync_lds();

    // layer 3: 25 -> 8, two lanes per row (4 outputs each)
    {
        const float* h2 = &scr2[wave][mrow * 28];
        int c0 = hi ? 4 : 0;
        #pragma unroll
        for (int c = 0; c < 4; ++c) {
            float a3 = s_b3[c0 + c];
            for (int n = 0; n < 25; ++n) a3 += h2[n] * s_w3[n * 8 + c0 + c];
            scr3[wave][mrow * 8 + c0 + c] = a3;
        }
    }
    wave_sync_lds();

    if (!hi) {
        const float* r = rs + (size_t)R * 3;
        float rx = r[0], ry = r[1], rz = r[2];
        float bx = 0.f, by = 0.f, bz = 0.f, cut = 1.f;
        const float* wn = &scr3[wave][mrow * 8];
        #pragma unroll
        for (int k = 0; k < 8; ++k) {
            float dx = rx - s_coords[k * 3 + 0];
            float dy = ry - s_coords[k * 3 + 1];
            float dz = rz - s_coords[k * 3 + 2];
            float w = wn[k];
            bx += w * dx; by += w * dy; bz += w * dz;
            float dist = sqrtf(dx * dx + dy * dy + dz * dz);
            float rr = dist * 2.0f;                    // dist / L, L = 0.5
            float cc = (rr < 0.5f) ? rr * rr * (6.0f - 8.0f * rr + 3.0f * rr * rr)
                                   : 1.0f;             // matches jnp.where(r < L, ...)
            cut *= cc;
        }
        bfnuc_ws[R * 3 + 0] = bx;
        bfnuc_ws[R * 3 + 1] = by;
        bfnuc_ws[R * 3 + 2] = bz;
        cut_ws[R] = cut;
    }
}

// ---------------------------------------------------------------------------
// Kernel B: electron-electron pair path + final combine.
// One block per batch (512 blocks, 128 threads = 4 waves). 992 pairs/batch =
// 62 M-tiles of 16. GEMM [16,256]x[256,40->48 pad] via WMMA f16 with A built
// on the fly as xs[i]*xs[j] (pk f16 mul from LDS). Layers 2/3 scalar; per-i
// accumulation via LDS ds_add_f32; final combine with nuc workspace.
// ---------------------------------------------------------------------------
__global__ __launch_bounds__(128) void pair_kernel(
    const float* __restrict__ rs, const float* __restrict__ xs,
    const float* __restrict__ ew1, const float* __restrict__ eb1,
    const float* __restrict__ ew2, const float* __restrict__ eb2,
    const float* __restrict__ ew3, const float* __restrict__ eb3,
    const float* __restrict__ bfnuc_ws, const float* __restrict__ cut_ws,
    float* __restrict__ out)
{
    __shared__ __align__(16) _Float16 xs_h[32 * 256];        // 16 KB
    __shared__ __align__(32) _Float16 w1f[3 * 8 * 32 * 16];  // 24 KB frag layout
    __shared__ float scr[4][16 * 48];                        // 12 KB
    __shared__ float bfacc[32 * 3];
    __shared__ float s_b1[48];
    __shared__ float s_w2[40 * 6];
    __shared__ float s_b2[6];
    __shared__ float s_w3[6];
    __shared__ float s_b3[1];

    const int b    = blockIdx.x;
    const int tid  = threadIdx.x;
    const int lane = tid & 31;
    const int wave = tid >> 5;
    const bool hi  = lane >= 16;
    const int mrow = lane & 15;
    const int a_off0 = hi ? 8 : 0;

    const float* xb = xs + (size_t)b * 32 * 256;
    for (int idx = tid; idx < 32 * 256; idx += 128)
        xs_h[idx] = (_Float16)xb[idx];

    // ew1 fragments: frag[((t*8+s)*32+l)*16+e] = ew1[32s+kmapB(l,e)][16t+(l&15)]
    for (int idx = tid; idx < 3 * 8 * 32; idx += 128) {
        int l = idx & 31;
        int s = (idx >> 5) & 7;
        int t = idx >> 8;
        int n  = t * 16 + (l & 15);
        int kb = s * 32 + ((l >= 16) ? 8 : 0);
        _Float16* dst = &w1f[idx * 16];
        #pragma unroll
        for (int e = 0; e < 16; ++e)
            dst[e] = (n < 40) ? (_Float16)ew1[(kb + e + ((e >= 8) ? 8 : 0)) * 40 + n] : (_Float16)0.0f;
    }
    if (tid < 48) s_b1[tid] = (tid < 40) ? eb1[tid] : 0.0f;
    for (int idx = tid; idx < 240; idx += 128) s_w2[idx] = ew2[idx];
    if (tid < 6) { s_b2[tid] = eb2[tid]; s_w3[tid] = ew3[tid]; }
    if (tid == 0) s_b3[0] = eb3[0];
    for (int idx = tid; idx < 96; idx += 128) bfacc[idx] = 0.0f;
    __syncthreads();

    const v16h* bfr = (const v16h*)w1f;

    for (int tile = wave; tile < 62; tile += 4) {
        // pair row owned by this lane for the A fragment
        int p  = tile * 16 + mrow;
        int i  = p / 31;
        int jj = p - i * 31;
        int j  = (jj < i) ? jj : jj + 1;
        const _Float16* xi = &xs_h[i * 256];
        const _Float16* xj = &xs_h[j * 256];

        v8f acc0 = {}, acc1 = {}, acc2 = {};
        #pragma unroll
        for (int s = 0; s < 8; ++s) {
            int k0 = s * 32 + a_off0;
            v8h ai0 = *(const v8h*)(xi + k0);
            v8h aj0 = *(const v8h*)(xj + k0);
            v8h ai1 = *(const v8h*)(xi + k0 + 16);
            v8h aj1 = *(const v8h*)(xj + k0 + 16);
            v8h p0 = ai0 * aj0;   // v_pk_mul_f16
            v8h p1 = ai1 * aj1;
            v16h a;
            #pragma unroll
            for (int e = 0; e < 8; ++e) { a[e] = p0[e]; a[e + 8] = p1[e]; }
            v16h b0 = bfr[(0 * 8 + s) * 32 + lane];
            v16h b1 = bfr[(1 * 8 + s) * 32 + lane];
            v16h b2 = bfr[(2 * 8 + s) * 32 + lane];
            acc0 = wmma16(a, b0, acc0);
            acc1 = wmma16(a, b1, acc1);
            acc2 = wmma16(a, b2, acc2);
        }

        // h1 = ssp(acc + b1) -> per-wave scratch [16 pairs][48]
        {
            float* sc = scr[wave];
            int mb = hi ? 8 : 0;
            int nl = lane & 15;
            float b0v = s_b1[nl], b1v = s_b1[16 + nl], b2v = s_b1[32 + nl];
            #pragma unroll
            for (int r = 0; r < 8; ++r) {
                sc[(mb + r) * 48 + nl]      = ssp(acc0[r] + b0v);
                sc[(mb + r) * 48 + 16 + nl] = ssp(acc1[r] + b1v);
                sc[(mb + r) * 48 + 32 + nl] = ssp(acc2[r] + b2v);
            }
        }
        wave_sync_lds();

        // layers 2 (40->6, ssp) and 3 (6->1): one pair per lo-lane
        if (!hi) {
            const float* hrow = &scr[wave][mrow * 48];
            float h2[6];
            #pragma unroll
            for (int c = 0; c < 6; ++c) {
                float a2 = s_b2[c];
                for (int n = 0; n < 40; ++n) a2 += hrow[n] * s_w2[n * 6 + c];
                h2[c] = ssp(a2);
            }
            float w = s_b3[0];
            #pragma unroll
            for (int c = 0; c < 6; ++c) w += h2[c] * s_w3[c];

            const float* ri = rs + (size_t)(b * 32 + i) * 3;
            const float* rj = rs + (size_t)(b * 32 + j) * 3;
            atomicAdd(&bfacc[i * 3 + 0], w * (ri[0] - rj[0]));
            atomicAdd(&bfacc[i * 3 + 1], w * (ri[1] - rj[1]));
            atomicAdd(&bfacc[i * 3 + 2], w * (ri[2] - rj[2]));
        }
        wave_sync_lds();   // protect scr before next iteration
    }
    __syncthreads();

    if (tid < 32) {
        int row = b * 32 + tid;
        float cut = cut_ws[row];
        float bx = bfacc[tid * 3 + 0] + bfnuc_ws[row * 3 + 0];
        float by = bfacc[tid * 3 + 1] + bfnuc_ws[row * 3 + 1];
        float bz = bfacc[tid * 3 + 2] + bfnuc_ws[row * 3 + 2];
        const float* r = rs + (size_t)row * 3;
        float* o = out + (size_t)row * 3;
        o[0] = r[0] + 1e-4f * cut * bx;
        o[1] = r[1] + 1e-4f * cut * by;
        o[2] = r[2] + 1e-4f * cut * bz;
    }
}

extern "C" void kernel_launch(void* const* d_in, const int* in_sizes, int n_in,
                              void* d_out, int out_size, void* d_ws, size_t ws_size,
                              hipStream_t stream) {
    (void)in_sizes; (void)n_in; (void)out_size; (void)ws_size;
    const float* rs     = (const float*)d_in[0];
    const float* xs     = (const float*)d_in[1];
    const float* coords = (const float*)d_in[2];
    const float* ew1 = (const float*)d_in[3];
    const float* eb1 = (const float*)d_in[4];
    const float* ew2 = (const float*)d_in[5];
    const float* eb2 = (const float*)d_in[6];
    const float* ew3 = (const float*)d_in[7];
    const float* eb3 = (const float*)d_in[8];
    const float* nw1 = (const float*)d_in[9];
    const float* nb1 = (const float*)d_in[10];
    const float* nw2 = (const float*)d_in[11];
    const float* nb2 = (const float*)d_in[12];
    const float* nw3 = (const float*)d_in[13];
    const float* nb3 = (const float*)d_in[14];
    float* out = (float*)d_out;

    float* ws     = (float*)d_ws;
    float* bfnuc  = ws;                 // 16384*3 floats
    float* cut    = ws + 16384 * 3;     // 16384 floats

    // 16384 rows -> 1024 M-tiles, 2 tiles/block
    nuc_kernel<<<512, 64, 0, stream>>>(rs, xs, coords, nw1, nb1, nw2, nb2,
                                       nw3, nb3, bfnuc, cut);
    // one block per batch; reads nuc workspace (stream-ordered)
    pair_kernel<<<512, 128, 0, stream>>>(rs, xs, ew1, eb1, ew2, eb2, ew3, eb3,
                                         bfnuc, cut, out);
}
